// FullyConnectedTensorProductRescaleNorm_25434796327210
// MI455X (gfx1250) — matrix operand order repaired and checked by reading, not compiled, not run on hardware
//
#include <hip/hip_runtime.h>
#include <hip/hip_bf16.h>

typedef __attribute__((ext_vector_type(16))) _Float16 v16h;
typedef __attribute__((ext_vector_type(8)))  _Float16 v8h;
typedef __attribute__((ext_vector_type(8)))  float    v8f;

#define MUL0 256
#define MUL1 128
#define DIM  640
#define NGRP 2000
#define BM   64          // rows per block in GEMM kernel
#define P_S1 264         // LDS pitch (halfs) for K=256 pieces (multiple of 8)
#define P_K1 136         // LDS pitch (halfs) for K=128 pieces (multiple of 8)
#define WHALF_TOTAL 163840   // total f16 weight elements (all 5 matrices)

// ---------------------------------------------------------------------------
// Kernel 1: convert f32 weights -> f16, pre-swizzled into WMMA B-tile layout.
// B tile is 32(K) x 16(N). Storage: tile*512 + lane*16 + e, where
// lane = n%16 + 16*(k/16 within tile), e = k%16 (pairs pack into VGPRs).
// Tiles laid out tileIdx = kt*numNT + nt.
// Scale folds: W110 *= 1/sqrt(3), W111 *= 1/sqrt(2).
// ---------------------------------------------------------------------------
__global__ __launch_bounds__(256) void prep_weights(
    const float* __restrict__ w000, const float* __restrict__ w110,
    const float* __restrict__ w011, const float* __restrict__ w101,
    const float* __restrict__ w111,
    _Float16* __restrict__ d000, _Float16* __restrict__ d110,
    _Float16* __restrict__ d011, _Float16* __restrict__ d101,
    _Float16* __restrict__ d111)
{
    int b = blockIdx.x;
    const float* src; _Float16* dst; int NC; int tile; float scale = 1.0f;
    if (b < 128)      { src = w000; dst = d000; NC = 256; tile = b; }
    else if (b < 192) { src = w110; dst = d110; NC = 256; tile = b - 128; scale = 0.57735026919f; }
    else if (b < 256) { src = w011; dst = d011; NC = 128; tile = b - 192; }
    else if (b < 288) { src = w101; dst = d101; NC = 128; tile = b - 256; }
    else              { src = w111; dst = d111; NC = 128; tile = b - 288; scale = 0.70710678119f; }
    int numNT = NC >> 4;
    int kt = tile / numNT, nt = tile % numNT;
    for (int ei = threadIdx.x; ei < 512; ei += 256) {
        int L  = ei >> 4;            // lane 0..31
        int e  = ei & 15;            // within-lane element
        int kk = ((L >> 4) << 4) + e;  // k within 32-K tile
        int nn = L & 15;
        int k = kt * 32 + kk, nc = nt * 16 + nn;
        dst[tile * 512 + ei] = (_Float16)(src[k * NC + nc] * scale);
    }
}

// ---------------------------------------------------------------------------
// Kernel 2: fused tensor-product GEMM via v_wmma_f32_16x16x32_f16.
// Block = 256 threads = 8 waves: 4 row-strips (16 rows) x 2 column halves.
// Covers BM=64 rows. A operands staged in LDS as f16 (row-major, K contiguous).
// ---------------------------------------------------------------------------
__global__ __launch_bounds__(256) void tp_gemm(
    const float* __restrict__ x, const float* __restrict__ y,
    const _Float16* __restrict__ wW000, const _Float16* __restrict__ wW110,
    const _Float16* __restrict__ wW011, const _Float16* __restrict__ wW101,
    const _Float16* __restrict__ wW111,
    const float* __restrict__ bias0,
    float* __restrict__ out0ws, float* __restrict__ out1ws, int n_total)
{
    __shared__ __align__(16) _Float16 sS1[BM * P_S1];       // s1 (K=256)
    __shared__ __align__(16) _Float16 sD [BM * P_K1];       // v1.v2 (K=128)
    __shared__ __align__(16) _Float16 sV [3][BM * P_K1];    // v1_k
    __shared__ __align__(16) _Float16 sC [3][BM * P_K1];    // cross_k
    __shared__ __align__(16) float    sY [BM][4];           // s2, v2xyz

    const int tid = threadIdx.x;
    const int rowBase = blockIdx.x * BM;

    // -------- prefetch f16 weight region into caches (contiguous 320KB) ----
    {
        const char* wb = (const char*)wW000;
        for (int off = tid * 256; off < WHALF_TOTAL * 2; off += 256 * 256)
            __builtin_prefetch(wb + off, 0, 1);
    }

    // -------- staging pass 1: load x,y -> f16 LDS --------------------------
    {
        int r = tid >> 2, q = tid & 3;          // 4 threads per row
        long n = rowBase + r;
        bool valid = n < n_total;
        const float4* xrow = (const float4*)(x + (long)n * DIM);
        // scalars: float4 idx 0..63
        for (int i = 0; i < 16; ++i) {
            int f4 = q * 16 + i;
            float4 v = valid ? xrow[f4] : float4{0.f, 0.f, 0.f, 0.f};
            int c = f4 * 4;
            sS1[r * P_S1 + c + 0] = (_Float16)v.x;
            sS1[r * P_S1 + c + 1] = (_Float16)v.y;
            sS1[r * P_S1 + c + 2] = (_Float16)v.z;
            sS1[r * P_S1 + c + 3] = (_Float16)v.w;
        }
        // vectors: float4 idx 64..159 (elements 0..383 of the v-block)
        for (int i = 0; i < 24; ++i) {
            int f4 = q * 24 + i;
            float4 v = valid ? xrow[64 + f4] : float4{0.f, 0.f, 0.f, 0.f};
            float fv[4] = {v.x, v.y, v.z, v.w};
            for (int j = 0; j < 4; ++j) {
                int e = f4 * 4 + j;             // 0..383
                int u = e / 3, k = e - 3 * u;
                sV[k][r * P_K1 + u] = (_Float16)fv[j];
            }
        }
        if (q == 0) {
            const float* yr = y + (long)n * 4;
            sY[r][0] = valid ? yr[0] : 0.f;
            sY[r][1] = valid ? yr[1] : 0.f;
            sY[r][2] = valid ? yr[2] : 0.f;
            sY[r][3] = valid ? yr[3] : 0.f;
        }
    }
    __syncthreads();

    // -------- staging pass 2: dot + cross products -------------------------
    for (int it = tid; it < BM * MUL1; it += 256) {
        int r = it >> 7, u = it & 127;
        float a0 = (float)sV[0][r * P_K1 + u];
        float a1 = (float)sV[1][r * P_K1 + u];
        float a2 = (float)sV[2][r * P_K1 + u];
        float b0 = sY[r][1], b1 = sY[r][2], b2 = sY[r][3];
        sD[r * P_K1 + u]    = (_Float16)(a0 * b0 + a1 * b1 + a2 * b2);
        sC[0][r * P_K1 + u] = (_Float16)(a1 * b2 - a2 * b1);
        sC[1][r * P_K1 + u] = (_Float16)(a2 * b0 - a0 * b2);
        sC[2][r * P_K1 + u] = (_Float16)(a0 * b1 - a1 * b0);
    }
    __syncthreads();

    // -------- WMMA phase ---------------------------------------------------
    const int lane = tid & 31, wid = tid >> 5;
    const int rw = wid & 3;        // row strip (16 rows)
    const int ch = wid >> 1 >> 1;  // column half (wid>>2)
    const int mrow = (rw << 4) + (lane & 15);
    const int kh = (lane >> 4) << 4;   // 0 or 16: k-offset by lane half

    auto loadA = [&](const _Float16* base, int pitch, int kt) -> v16h {
        const _Float16* p = base + mrow * pitch + kt * 32 + kh;
        v8h lo = *(const v8h*)p;            // ds_load_b128
        v8h hi = *(const v8h*)(p + 8);      // ds_load_b128
        v16h a;
        #pragma unroll
        for (int i = 0; i < 8; ++i) { a[i] = lo[i]; a[8 + i] = hi[i]; }
        return a;
    };
    auto loadB = [&](const _Float16* wbase, int tile) -> v16h {
        const _Float16* p = wbase + tile * 512 + lane * 16;
        v8h lo = *(const v8h*)p;            // global_load_b128
        v8h hi = *(const v8h*)(p + 8);      // global_load_b128
        v16h b;
        #pragma unroll
        for (int i = 0; i < 8; ++i) { b[i] = lo[i]; b[8 + i] = hi[i]; }
        return b;
    };
    auto mma = [&](v16h a, v16h b, v8f c) -> v8f {
        return __builtin_amdgcn_wmma_f32_16x16x32_f16(
            false, a, false, b, (short)0, c, false, false);
    };

    // per-row scalars for the 8 C rows this lane owns
    float s2v[8], v2v[3][8];
    #pragma unroll
    for (int i = 0; i < 8; ++i) {
        int rr = (rw << 4) + ((lane >> 4) << 3) + i;
        float4 yv = *(const float4*)(&sY[rr][0]);
        s2v[i] = yv.x; v2v[0][i] = yv.y; v2v[1][i] = yv.z; v2v[2][i] = yv.w;
    }

    const v8f vzero = {};
    const int rbase = rowBase + (rw << 4) + ((lane >> 4) << 3);
    const int nlo = lane & 15;

    // ===== out0: (s1@W000)*s2 + d@W110' + bias0 (wave: 8 col tiles) ========
    {
        v8f acc[8];
        #pragma unroll
        for (int j = 0; j < 8; ++j) acc[j] = vzero;
        for (int kt = 0; kt < 8; ++kt) {
            v16h a = loadA(sS1, P_S1, kt);
            #pragma unroll
            for (int j = 0; j < 8; ++j)
                acc[j] = mma(a, loadB(wW000, kt * 16 + ch * 8 + j), acc[j]);
        }
        #pragma unroll
        for (int j = 0; j < 8; ++j)
            #pragma unroll
            for (int i = 0; i < 8; ++i) acc[j][i] *= s2v[i];
        for (int kt = 0; kt < 4; ++kt) {
            v16h a = loadA(sD, P_K1, kt);
            #pragma unroll
            for (int j = 0; j < 8; ++j)
                acc[j] = mma(a, loadB(wW110, kt * 16 + ch * 8 + j), acc[j]);
        }
        #pragma unroll
        for (int j = 0; j < 8; ++j) {
            int col = ch * 128 + j * 16 + nlo;
            float bb = bias0[col];
            #pragma unroll
            for (int i = 0; i < 8; ++i) {
                long n = rbase + i;
                if (n < n_total) out0ws[n * MUL0 + col] = acc[j][i] + bb;
            }
        }
    }

    // ===== t = s1@W011 (wave: 4 col tiles of the 128-wide space) ===========
    v8f tacc[4];
    #pragma unroll
    for (int j = 0; j < 4; ++j) tacc[j] = vzero;
    for (int kt = 0; kt < 8; ++kt) {
        v16h a = loadA(sS1, P_S1, kt);
        #pragma unroll
        for (int j = 0; j < 4; ++j)
            tacc[j] = mma(a, loadB(wW011, kt * 8 + ch * 4 + j), tacc[j]);
    }

    // ===== out1_k = (v1k@W101)*s2 + crossk@W111' + t*v2k ===================
    for (int k = 0; k < 3; ++k) {
        v8f acc[4];
        #pragma unroll
        for (int j = 0; j < 4; ++j) acc[j] = vzero;
        for (int kt = 0; kt < 4; ++kt) {
            v16h a = loadA(&sV[k][0], P_K1, kt);
            #pragma unroll
            for (int j = 0; j < 4; ++j)
                acc[j] = mma(a, loadB(wW101, kt * 8 + ch * 4 + j), acc[j]);
        }
        #pragma unroll
        for (int j = 0; j < 4; ++j)
            #pragma unroll
            for (int i = 0; i < 8; ++i) acc[j][i] *= s2v[i];
        for (int kt = 0; kt < 4; ++kt) {
            v16h a = loadA(&sC[k][0], P_K1, kt);
            #pragma unroll
            for (int j = 0; j < 4; ++j)
                acc[j] = mma(a, loadB(wW111, kt * 8 + ch * 4 + j), acc[j]);
        }
        float* plane = out1ws + (long)k * n_total * MUL1;
        #pragma unroll
        for (int j = 0; j < 4; ++j) {
            int col = ch * 64 + j * 16 + nlo;
            #pragma unroll
            for (int i = 0; i < 8; ++i) {
                long n = rbase + i;
                float v = acc[j][i] + tacc[j][i] * v2v[k][i];
                if (n < n_total) plane[n * MUL1 + col] = v;
            }
        }
    }
}

// ---------------------------------------------------------------------------
// Kernel 3: per-segment statistics (batch is sorted -> binary search range).
// stats[g*640 + 0..255]  = mu*mean_shift (shift)
// stats[g*640 + 256..511]= rsqrt(E[f0^2]+eps)*aw[:256]
// stats[g*640 + 512..639]= rsqrt(E[|out1|^2/3]+eps)*aw[256:]
// ---------------------------------------------------------------------------
__global__ __launch_bounds__(384) void seg_stats(
    const float* __restrict__ out0ws, const float* __restrict__ out1ws,
    const int* __restrict__ batch,
    const float* __restrict__ mean_shift, const float* __restrict__ affine_weight,
    float* __restrict__ stats, int n_total)
{
    int g = blockIdx.x;
    int lo = 0, hi = n_total;
    while (lo < hi) { int m = (lo + hi) >> 1; if (batch[m] < g) lo = m + 1; else hi = m; }
    int lo2 = lo, hi2 = n_total;
    while (lo2 < hi2) { int m = (lo2 + hi2) >> 1; if (batch[m] < g + 1) lo2 = m + 1; else hi2 = m; }
    int beg = lo, end = lo2;
    float cnt = fmaxf((float)(end - beg), 1.0f);
    int t = threadIdx.x;
    if (t < 256) {
        float s = 0.f, ss = 0.f;
        for (int n = beg; n < end; ++n) {
            float v = out0ws[(long)n * MUL0 + t];
            s += v; ss += v * v;
        }
        float mu = s / cnt;
        float ms = mean_shift[t];
        float ef2 = ss / cnt - mu * mu * ms * (2.0f - ms);
        stats[(long)g * 640 + t]       = mu * ms;
        stats[(long)g * 640 + 256 + t] = rsqrtf(ef2 + 1e-5f) * affine_weight[t];
    } else {
        int w = t - 256;
        long plane = (long)n_total * MUL1;
        float q = 0.f;
        for (int n = beg; n < end; ++n) {
            float a = out1ws[(long)n * MUL1 + w];
            float b = out1ws[plane + (long)n * MUL1 + w];
            float c = out1ws[2 * plane + (long)n * MUL1 + w];
            q += a * a + b * b + c * c;
        }
        stats[(long)g * 640 + 512 + w] =
            rsqrtf(q / (3.0f * cnt) + 1e-5f) * affine_weight[256 + w];
    }
}

// ---------------------------------------------------------------------------
// Kernel 4: apply normalization, write concatenated (N,640) output.
// ---------------------------------------------------------------------------
__global__ __launch_bounds__(384) void seg_apply(
    const float* __restrict__ out0ws, const float* __restrict__ out1ws,
    const float* __restrict__ stats, const int* __restrict__ batch,
    const float* __restrict__ affine_bias, float* __restrict__ out, int n_total)
{
    int n = blockIdx.x;
    if (n >= n_total) return;
    const float* st = stats + (long)batch[n] * 640;
    int t = threadIdx.x;
    if (t < 256) {
        float v = out0ws[(long)n * MUL0 + t];
        out[(long)n * DIM + t] = (v - st[t]) * st[256 + t] + affine_bias[t];
    } else {
        int w = t - 256;
        long plane = (long)n_total * MUL1;
        float n1 = st[512 + w];
        float a = out1ws[(long)n * MUL1 + w];
        float b = out1ws[plane + (long)n * MUL1 + w];
        float c = out1ws[2 * plane + (long)n * MUL1 + w];
        float* o = out + (long)n * DIM + MUL0 + 3 * w;
        o[0] = a * n1; o[1] = b * n1; o[2] = c * n1;
    }
}

// ---------------------------------------------------------------------------
extern "C" void kernel_launch(void* const* d_in, const int* in_sizes, int n_in,
                              void* d_out, int out_size, void* d_ws, size_t ws_size,
                              hipStream_t stream) {
    const float* x             = (const float*)d_in[0];
    const float* y             = (const float*)d_in[1];
    const float* w000          = (const float*)d_in[2];
    const float* w011          = (const float*)d_in[3];
    const float* w101          = (const float*)d_in[4];
    const float* w110          = (const float*)d_in[5];
    const float* w111          = (const float*)d_in[6];
    const float* bias0         = (const float*)d_in[7];
    const float* mean_shift    = (const float*)d_in[8];
    const float* affine_weight = (const float*)d_in[9];
    const float* affine_bias   = (const float*)d_in[10];
    const int*   batch         = (const int*)d_in[11];

    int n = in_sizes[0] / DIM;

    // workspace layout
    float* out0ws = (float*)d_ws;                       // n*256 f32
    float* out1ws = out0ws + (long)n * MUL0;            // 3 * n*128 f32
    _Float16* wh  = (_Float16*)(out1ws + (long)n * MUL1 * 3);
    _Float16* d000 = wh;                 // 65536
    _Float16* d110 = wh + 65536;         // 32768
    _Float16* d011 = wh + 98304;         // 32768
    _Float16* d101 = wh + 131072;        // 16384
    _Float16* d111 = wh + 147456;        // 16384
    float* stats  = (float*)(wh + WHALF_TOTAL);         // G*640 f32

    prep_weights<<<320, 256, 0, stream>>>(w000, w110, w011, w101, w111,
                                          d000, d110, d011, d101, d111);
    int nb = (n + BM - 1) / BM;
    tp_gemm<<<nb, 256, 0, stream>>>(x, y, d000, d110, d011, d101, d111,
                                    bias0, out0ws, out1ws, n);
    seg_stats<<<NGRP, 384, 0, stream>>>(out0ws, out1ws, batch,
                                        mean_shift, affine_weight, stats, n);
    seg_apply<<<n, 384, 0, stream>>>(out0ws, out1ws, stats, batch,
                                     affine_bias, (float*)d_out, n);
}
